// MAB_51857435131932
// MI455X (gfx1250) — matrix operand order repaired and checked by reading, not compiled, not run on hardware
//
#include <hip/hip_runtime.h>
#include <hip/hip_bf16.h>
#include <math.h>

// ---- problem constants (from reference) ----
#define B_SZ   8
#define NQ     1024
#define NK     1024
#define DIMV   512
#define HEADS  8
#define DHEAD  64          // DIMV / HEADS
#define OW     1024        // 2*DIMV, projection width

typedef __attribute__((ext_vector_type(16))) __bf16 v16bf;
typedef __attribute__((ext_vector_type(8)))  float  v8f;

// gfx1250 async global->LDS copy path (ASYNCcnt-tracked), guarded so the
// kernel still compiles if this toolchain doesn't declare the builtins.
#if defined(__has_builtin)
#if __has_builtin(__builtin_amdgcn_global_load_async_to_lds_b128) && \
    __has_builtin(__builtin_amdgcn_s_wait_asynccnt)
#define HAVE_ASYNC_LDS 1
#endif
#endif
#ifndef HAVE_ASYNC_LDS
#define HAVE_ASYNC_LDS 0
#endif

// Builtin signature (from hipcc diagnostic): param0 = v4i in AS1 (global,
// printed as "__device__"), param1 = v4i in AS3 (LDS), then imm offset, imm cpol.
typedef int v4i __attribute__((vector_size(16)));
typedef __attribute__((address_space(1))) v4i* gptr_v4i;
typedef __attribute__((address_space(3))) v4i* lptr_v4i;

// CDNA5 wave32 WMMA fragment index helpers (ISA 7.12.2):
//  A (16-bit, 16x32): lane L -> M = L&15, half = L>>4;
//    element e (0..15): K = half*16 + e (contiguous order chosen here, not the ISA table order)
//  B (16-bit, 32x16): lane L -> N = L&15, half = L>>4; element e: K = half*16 + e (same contiguous order as A)
//  C/D (f32, 16x16): lane L -> N = L&15; vgpr r -> M = r + 8*(L>>4)
__device__ __forceinline__ int a_kidx(int e, int half) {
    return half * 16 + e;
}

// ============================================================
// Kernel 1: attention for one (b, h, 16-q-row) tile per wave32.
// Emits: A1 (normalized probs) and O-scratch ([M | div] layout [B*NQ, 1024]).
// ============================================================
__global__ __launch_bounds__(32) void mab_attn_kernel(const float* __restrict__ Q,
                                                      const float* __restrict__ K,
                                                      float* __restrict__ Oscratch,
                                                      float* __restrict__ A1out) {
    const int tile = blockIdx.x;              // 0 .. B*HEADS*(NQ/16)-1
    const int qt   = tile % (NQ / 16);
    const int bh   = tile / (NQ / 16);
    const int h    = bh % HEADS;
    const int b    = bh / HEADS;
    const int q0   = qt * 16;

    const int lane = threadIdx.x;             // 0..31 (wave32)
    const int half = lane >> 4;
    const int mrow = lane & 15;               // A-row / B-col / C-col index

    __shared__ float s_scores[16 * NK];       // 64 KB: raw scores -> probs
    __shared__ float s_kstage[32 * DHEAD];    // 8 KB: staged V chunk (32 keys x 64 dims)

    const float scale = 0.0441941738241592f;  // 1/sqrt(512) (note: dim_V, not d_head)

    const float* Qbase = Q + ((size_t)b * NQ + q0) * DIMV + h * DHEAD;
    const float* Kbase = K + (size_t)b * NK * DIMV + h * DHEAD;

    // ---- Q A-fragments: 16x64 bf16 as two 16x32 chunks ----
    v16bf qfrag[2];
#pragma unroll
    for (int c = 0; c < 2; ++c) {
#pragma unroll
        for (int e = 0; e < 16; ++e) {
            int kd = c * 32 + a_kidx(e, half);
            qfrag[c][e] = (__bf16)Qbase[(size_t)mrow * DIMV + kd];
        }
    }

    // ---- pass 1: scores = Q * K^T for all 1024 keys ----
    const v8f vzero = {0.f, 0.f, 0.f, 0.f, 0.f, 0.f, 0.f, 0.f};
    for (int nt = 0; nt < NK / 16; ++nt) {
        const float* Kt = Kbase + (size_t)nt * 16 * DIMV;
        // prefetch the next key tile (global_prefetch_b8; speculative, drop-on-fault)
        __builtin_prefetch(Kt + 16 * DIMV, 0, 1);

        v8f acc = vzero;
#pragma unroll
        for (int c = 0; c < 2; ++c) {
            // B = K^T chunk: B[kd][n] = Ktile[key=n][dim=c*32+kd]
            v16bf bfrag;
#pragma unroll
            for (int e = 0; e < 16; ++e) {
                int kd = c * 32 + half * 16 + e;
                bfrag[e] = (__bf16)Kt[(size_t)mrow * DIMV + kd];
            }
            acc = __builtin_amdgcn_wmma_f32_16x16x32_bf16(
                false, qfrag[c], false, bfrag, (short)0, acc, false, false);
        }
        // scatter C tile into LDS score slab
#pragma unroll
        for (int r = 0; r < 8; ++r) {
            int M = r + 8 * half;
            s_scores[M * NK + nt * 16 + mrow] = acc[r];
        }
    }
    __syncthreads();

    // ---- softmax over each of the 16 rows (2 lanes per row) ----
    float* srow = &s_scores[mrow * NK];
    const int k0 = half * (NK / 2), k1 = k0 + (NK / 2);

    float lmax = -3.4e38f;
    for (int k = k0; k < k1; ++k) lmax = __builtin_fmaxf(lmax, srow[k]);
    float rmax = __builtin_fmaxf(lmax, __shfl_xor(lmax, 16, 32)) * scale;

    float lsum = 0.f;
    for (int k = k0; k < k1; ++k) {
        float e = __expf(srow[k] * scale - rmax);
        srow[k] = e;
        lsum += e;
    }
    float rinv = 1.0f / (lsum + __shfl_xor(lsum, 16, 32));

    // normalize in LDS and stream A1 to HBM: A1[b][q][h*NK + k]
    float* A1row = A1out + ((size_t)b * NQ + q0 + mrow) * ((size_t)HEADS * NK) + (size_t)h * NK;
    for (int k = k0; k < k1; ++k) {
        float p = srow[k] * rinv;
        srow[k] = p;
        A1row[k] = p;
    }
    __syncthreads();

    // ---- pass 2: M = A*V, M2 = A*V^2 (V = K) ----
    v8f accM[4] = {vzero, vzero, vzero, vzero};
    v8f accS[4] = {vzero, vzero, vzero, vzero};
    for (int kc = 0; kc < NK / 32; ++kc) {
        const float* Kt = Kbase + (size_t)kc * 32 * DIMV;
        __builtin_prefetch(Kt + 32 * DIMV, 0, 1);

        // stage 32 keys x 64 dims (8 KB) into LDS, one float4 per lane per step
#pragma unroll
        for (int i = 0; i < 16; ++i) {
            int j    = i * 32 + lane;           // float4 index 0..511
            int fidx = j * 4;                   // flat float index
            int key  = fidx >> 6;               // /64 dims
            int dim  = fidx & 63;
            const float* src = Kt + (size_t)key * DIMV + dim;
#if HAVE_ASYNC_LDS
            __builtin_amdgcn_global_load_async_to_lds_b128(
                (gptr_v4i)(void*)const_cast<float*>(src),
                (lptr_v4i)(void*)&s_kstage[fidx], 0, 0);
#else
            *(float4*)(&s_kstage[fidx]) = *(const float4*)src;
#endif
        }
#if HAVE_ASYNC_LDS
        __builtin_amdgcn_s_wait_asynccnt(0);
#endif
        __syncthreads();

        // A-fragment from normalized probs in LDS (f32 -> bf16)
        v16bf afrag;
#pragma unroll
        for (int e = 0; e < 16; ++e) {
            int kk = kc * 32 + a_kidx(e, half);
            afrag[e] = (__bf16)s_scores[mrow * NK + kk];
        }
#pragma unroll
        for (int nt = 0; nt < 4; ++nt) {
            v16bf bfrag, b2frag;
#pragma unroll
            for (int e = 0; e < 16; ++e) {
                int key = half * 16 + e;                       // key within chunk
                float v = s_kstage[key * DHEAD + nt * 16 + mrow];
                bfrag[e]  = (__bf16)v;
                b2frag[e] = (__bf16)(v * v);
            }
            accM[nt] = __builtin_amdgcn_wmma_f32_16x16x32_bf16(
                false, afrag, false, bfrag, (short)0, accM[nt], false, false);
            accS[nt] = __builtin_amdgcn_wmma_f32_16x16x32_bf16(
                false, afrag, false, b2frag, (short)0, accS[nt], false, false);
        }
        __syncthreads();   // protect s_kstage before next chunk overwrites it
    }

    // ---- epilogue: O-scratch rows [M(512) | div(512)], col = h*64 + dim ----
    float* Orow0 = Oscratch + ((size_t)b * NQ + q0) * OW;
#pragma unroll
    for (int nt = 0; nt < 4; ++nt) {
#pragma unroll
        for (int r = 0; r < 8; ++r) {
            int M = r + 8 * half;
            float m  = accM[nt][r];
            float m2 = accS[nt][r];
            float dv = sqrtf(__builtin_fmaxf(m2 - m * m, 1e-6f));
            size_t col = (size_t)h * DHEAD + nt * 16 + mrow;
            Orow0[(size_t)M * OW + col]        = m;
            Orow0[(size_t)M * OW + DIMV + col] = dv;
        }
    }
}

// ============================================================
// Kernel 2: O_out = O + relu(O @ W^T + bias).  One wave32 per
// 16-row x 64-col output tile; GEMM is [8192 x 1024] * [1024 x 1024].
// ============================================================
__global__ __launch_bounds__(32) void mab_proj_kernel(const float* __restrict__ Osc,
                                                      const float* __restrict__ W,
                                                      const float* __restrict__ bias,
                                                      float* __restrict__ Out) {
    const int mt = blockIdx.x;     // 0..511  (16 rows each, B*NQ = 8192 rows)
    const int ng = blockIdx.y;     // 0..15   (64 cols each)
    const int lane = threadIdx.x;
    const int half = lane >> 4;
    const int mrow = lane & 15;
    const int m0 = mt * 16;

    const float* Arow = Osc + (size_t)m0 * OW;
    const v8f vzero = {0.f, 0.f, 0.f, 0.f, 0.f, 0.f, 0.f, 0.f};
    v8f acc[4] = {vzero, vzero, vzero, vzero};

    for (int kc = 0; kc < OW / 32; ++kc) {
        v16bf afrag;
#pragma unroll
        for (int e = 0; e < 16; ++e) {
            int k = kc * 32 + a_kidx(e, half);
            afrag[e] = (__bf16)Arow[(size_t)mrow * OW + k];
        }
#pragma unroll
        for (int nt = 0; nt < 4; ++nt) {
            int n = ng * 64 + nt * 16 + mrow;     // B[k][n] = W[n][k] (O @ W^T)
            v16bf bfrag;
#pragma unroll
            for (int e = 0; e < 16; ++e) {
                int k = kc * 32 + half * 16 + e;
                bfrag[e] = (__bf16)W[(size_t)n * OW + k];
            }
            acc[nt] = __builtin_amdgcn_wmma_f32_16x16x32_bf16(
                false, afrag, false, bfrag, (short)0, acc[nt], false, false);
        }
    }

    // epilogue: out = O + relu(acc + bias)
#pragma unroll
    for (int nt = 0; nt < 4; ++nt) {
        int n = ng * 64 + nt * 16 + mrow;
        float bn = bias[n];
#pragma unroll
        for (int r = 0; r < 8; ++r) {
            int M = m0 + r + 8 * half;
            float o = Osc[(size_t)M * OW + n];
            float v = acc[nt][r] + bn;
            v = v > 0.f ? v : 0.f;
            Out[(size_t)M * OW + n] = o + v;
        }
    }
}

// ============================================================
// launch: inputs in setup_inputs order: Q, K, W_o, b_o.
// d_out = [O (8*1024*1024 f32) | A1 (8*1024*8192 f32)].
// d_ws  = O-scratch, 8192*1024 f32 = 32 MB.
// ============================================================
extern "C" void kernel_launch(void* const* d_in, const int* in_sizes, int n_in,
                              void* d_out, int out_size, void* d_ws, size_t ws_size,
                              hipStream_t stream) {
    const float* Q  = (const float*)d_in[0];
    const float* K  = (const float*)d_in[1];
    const float* Wo = (const float*)d_in[2];
    const float* bo = (const float*)d_in[3];

    float* out = (float*)d_out;
    float* A1  = out + (size_t)B_SZ * NQ * OW;     // after O block
    float* Osc = (float*)d_ws;                     // [B*NQ, 1024] f32

    dim3 g1(B_SZ * HEADS * (NQ / 16));             // 4096 waves
    mab_attn_kernel<<<g1, 32, 0, stream>>>(Q, K, Osc, A1);

    dim3 g2((B_SZ * NQ) / 16, OW / 64);            // 512 x 16 waves
    mab_proj_kernel<<<g2, 32, 0, stream>>>(Osc, Wo, bo, out);
}
